// MultiHeadedSelfRetentionModule_40638980555015
// MI455X (gfx1250) — matrix operand order repaired and checked by reading, not compiled, not run on hardware
//
#include <hip/hip_runtime.h>
#include <math.h>

// MultiHeadedSelfRetention forward for MI455X (gfx1250, wave32, WMMA bf16).
// Pipeline: LN -> fused QKVG GEMM (WMMA bf16) -> theta-shift -> per-chunk KV
// (WMMA, async-LDS staged) -> 8-step scan -> retention main (qk/inner/cross
// all WMMA, async-LDS staged v) -> RMS+silu -> output GEMM (WMMA).

#define DEV __device__ __forceinline__

typedef __attribute__((ext_vector_type(16))) __bf16 bf16x16;
typedef __attribute__((ext_vector_type(8)))  float  f32x8;

union Frag {
  bf16x16 v;
  uint4 q[2];
  unsigned short u[16];
};

DEV unsigned short f2bf(float f) {
  unsigned u = __float_as_uint(f);
  u += 0x7fffu + ((u >> 16) & 1u);   // round-to-nearest-even
  return (unsigned short)(u >> 16);
}
DEV float bf2f(unsigned short h) { return __uint_as_float(((unsigned)h) << 16); }

DEV f32x8 fzero() {
  f32x8 z;
  #pragma unroll
  for (int i = 0; i < 8; ++i) z[i] = 0.f;
  return z;
}

DEV bf16x16 zfrag() {
  Frag f;
  #pragma unroll
  for (int i = 0; i < 16; ++i) f.u[i] = 0;
  return f.v;
}

// CDNA5 async memory->LDS copy (16B), tracked by ASYNCcnt.
DEV void async_g2l_b128(unsigned short* ldsdst, const unsigned short* gsrc) {
  unsigned lds = (unsigned)(uintptr_t)ldsdst;     // low 32 bits = LDS offset
  asm volatile("global_load_async_to_lds_b128 %0, %1, off"
               :: "v"(lds), "v"(gsrc)
               : "memory");
}
DEV void wait_async0() {
  asm volatile("s_wait_asynccnt 0x0" ::: "memory");
}
DEV void zero16_lds(unsigned short* ldsdst) {
  uint4 z;
  z.x = z.y = z.z = z.w = 0u;
  *reinterpret_cast<uint4*>(ldsdst) = z;
}

// A operand fragment (16x32, M=lane%16, K map: e + 16*half -> 16 contiguous)
DEV bf16x16 loadA_row(const unsigned short* rowp, int k0, int half) {
  Frag f;
  f.q[0] = *reinterpret_cast<const uint4*>(rowp + k0 + 16 * half);
  f.q[1] = *reinterpret_cast<const uint4*>(rowp + k0 + 16 * half + 8);
  return f.v;
}
// B operand fragment (32x16, N=lane%16, K map: e + 16*half -> 16 contiguous)
DEV bf16x16 loadB_row(const unsigned short* rowp, int k0, int half) {
  Frag f;
  const uint4* p = reinterpret_cast<const uint4*>(rowp + k0 + 16 * half);
  f.q[0] = p[0];
  f.q[1] = p[1];
  return f.v;
}
// B fragment from row-major [K][ld] storage (strided by ld)
DEV bf16x16 loadB_strided(const unsigned short* base, int col, int k0, int half, int ld) {
  Frag f;
  #pragma unroll
  for (int e = 0; e < 16; ++e) {
    int k = k0 + e + 16 * half;
    f.u[e] = base[k * ld + col];
  }
  return f.v;
}
// A fragment where A[m][k] = base[k*ld + m] (transposed access)
DEV bf16x16 loadA_strided(const unsigned short* base, int mcol, int k0, int half, int ld) {
  Frag f;
  #pragma unroll
  for (int e = 0; e < 16; ++e) {
    int k = k0 + e + 16 * half;
    f.u[e] = base[k * ld + mcol];
  }
  return f.v;
}

DEV f32x8 wmma_bf16(bf16x16 a, bf16x16 b, f32x8 c) {
  return __builtin_amdgcn_wmma_f32_16x16x32_bf16(false, a, false, b, (short)0, c,
                                                 false, false);
}

DEV float decay_log(int h) { return logf(1.0f - exp2f(-5.0f - (float)h)); }

// ---------------- K0: weights -> bf16 transposed [5][N][K] ----------------
__global__ __launch_bounds__(256) void k_prep_w(const float* Wq, const float* Wk,
                                                const float* Wv, const float* Wg,
                                                const float* Wo, unsigned short* wt) {
  int idx = blockIdx.x * 256 + threadIdx.x;      // 5*2^20 total
  int w = idx >> 20;
  int rem = idx & 0xFFFFF;
  int nn = rem >> 10, kk = rem & 1023;
  const float* W = (w == 0) ? Wq : (w == 1) ? Wk : (w == 2) ? Wv : (w == 3) ? Wg : Wo;
  wt[idx] = f2bf(W[kk * 1024 + nn]);
}

// ---------------- K1: LayerNorm -> x bf16 ----------------
__global__ __launch_bounds__(256) void k_ln(const float* x, const float* gma,
                                            const float* bta, unsigned short* xb) {
  int row = blockIdx.x, tid = threadIdx.x;
  const float* xr = x + (size_t)row * 1024;
  float v[4];
  #pragma unroll
  for (int i = 0; i < 4; ++i) v[i] = xr[tid * 4 + i];
  __shared__ float red[256];
  float s = v[0] + v[1] + v[2] + v[3];
  red[tid] = s;
  __syncthreads();
  for (int off = 128; off; off >>= 1) {
    if (tid < off) red[tid] += red[tid + off];
    __syncthreads();
  }
  float mu = red[0] * (1.f / 1024.f);
  __syncthreads();
  float ss = 0.f;
  #pragma unroll
  for (int i = 0; i < 4; ++i) { float d = v[i] - mu; ss += d * d; }
  red[tid] = ss;
  __syncthreads();
  for (int off = 128; off; off >>= 1) {
    if (tid < off) red[tid] += red[tid + off];
    __syncthreads();
  }
  float inv = rsqrtf(red[0] * (1.f / 1024.f) + 1e-5f);
  #pragma unroll
  for (int i = 0; i < 4; ++i) {
    int d = tid * 4 + i;
    xb[(size_t)row * 1024 + d] = f2bf((v[i] - mu) * inv * gma[d] + bta[d]);
  }
}

// ---------------- K2: fused QKVG GEMM (WMMA) ----------------
__global__ __launch_bounds__(256) void k_gemm_qkvg(const unsigned short* xb,
                                                   const unsigned short* wt,
                                                   float* qf, float* kf,
                                                   unsigned short* vbb, float* gf) {
  int wave = threadIdx.x >> 5, lane = threadIdx.x & 31;
  int half = lane >> 4, ll = lane & 15;
  int row0 = blockIdx.x * 128 + wave * 16;
  const unsigned short* arow = xb + (size_t)(row0 + ll) * 1024;
  int w = blockIdx.y >> 3;       // which weight (q,k,v,g)
  int hcol = blockIdx.y & 7;     // head / col block within a weight
  const unsigned short* wbase = wt + ((size_t)blockIdx.y * 128) * 1024;
  f32x8 acc[8];
  #pragma unroll
  for (int i = 0; i < 8; ++i) acc[i] = fzero();
  for (int k0 = 0; k0 < 1024; k0 += 32) {
    if (k0 + 64 < 1024) {       // two k-steps ahead, WGP scope (locality 3)
      __builtin_prefetch(arow + k0 + 64, 0, 3);
      __builtin_prefetch(wbase + (size_t)ll * 1024 + k0 + 64, 0, 3);
    }
    bf16x16 a = loadA_row(arow, k0, half);
    #pragma unroll
    for (int nt = 0; nt < 8; ++nt) {
      const unsigned short* brow = wbase + (size_t)(nt * 16 + ll) * 1024;
      bf16x16 b = loadB_row(brow, k0, half);
      acc[nt] = wmma_bf16(a, b, acc[nt]);
    }
  }
  #pragma unroll
  for (int nt = 0; nt < 8; ++nt) {
    #pragma unroll
    for (int r = 0; r < 8; ++r) {
      int row = row0 + r + 8 * half;
      int b = row / 4000, t = row % 4000;
      int dk = nt * 16 + ll;
      float val = acc[nt][r];
      size_t hm = (((size_t)(b * 8 + hcol) * 4000 + t) * 128) + dk;  // head-major
      if (w == 0)      qf[hm] = val;
      else if (w == 1) kf[hm] = val * 0.08838834764831845f;          // DK^-0.5
      else if (w == 2) vbb[hm] = f2bf(val);
      else             gf[(size_t)row * 1024 + hcol * 128 + dk] = val;
    }
  }
}

// ---------------- K3: theta shift (RoPE-like) -> qr,kr bf16 ----------------
__global__ __launch_bounds__(256) void k_theta(const float* qf, const float* kf,
                                               unsigned short* qrb,
                                               unsigned short* krb) {
  int idx = blockIdx.x * 256 + threadIdx.x;   // B*H*T*64 pairs
  int j = idx & 63;
  int rest = idx >> 6;                        // bh*T + t
  int t = rest % 4000;
  float ang = (float)t * powf(10000.f, -(float)j * (1.f / 63.f));
  float s = sinf(ang), c = cosf(ang);
  size_t base = (size_t)rest * 128 + 2 * j;
  float q0 = qf[base], q1 = qf[base + 1];
  qrb[base]     = f2bf(q0 * c - q1 * s);
  qrb[base + 1] = f2bf(q1 * c + q0 * s);
  float k0 = kf[base], k1 = kf[base + 1];
  krb[base]     = f2bf(k0 * c - k1 * s);
  krb[base + 1] = f2bf(k1 * c + k0 * s);
}

// ---------------- K4: per-chunk kv_i = (kr*vid)^T @ v (WMMA) ----------------
// value_inner_decay is a per-row (c) factor, so it is folded into the kr
// staging (compute) while v is staged with pure async global->LDS DMA.
__global__ __launch_bounds__(256) void k_kv(const unsigned short* krb,
                                            const unsigned short* vbb, float* kvb) {
  int blk = blockIdx.x;           // bh*8 + n
  int n = blk & 7, bh = blk >> 3, h = bh & 7;
  int tid = threadIdx.x, wave = tid >> 5, lane = tid & 31;
  int half = lane >> 4, ll = lane & 15;
  float dh = decay_log(h);
  float lrs = (1.f - expf(dh * 500.f)) / (1.f - expf(dh));
  const unsigned short* krc = krb + ((size_t)bh * 4000 + n * 500) * 128;
  const unsigned short* vc  = vbb + ((size_t)bh * 4000 + n * 500) * 128;
  __shared__ unsigned short krsh[64 * 128];
  __shared__ unsigned short vsh[64 * 128];
  f32x8 acc[8];
  #pragma unroll
  for (int i = 0; i < 8; ++i) acc[i] = fzero();
  int m0 = wave * 16;
  for (int slab = 0; slab < 8; ++slab) {
    __syncthreads();
    // v rows: async DMA to LDS (16B chunks; 16 chunks per 128-col row)
    #pragma unroll
    for (int i = 0; i < 4; ++i) {
      int c16 = tid + 256 * i;            // 0..1023
      int lr = c16 >> 4, cc = (c16 & 15) * 8;
      int cr = slab * 64 + lr;
      if (cr < 500)
        async_g2l_b128(&vsh[lr * 128 + cc], vc + (size_t)cr * 128 + cc);
      else
        zero16_lds(&vsh[lr * 128 + cc]);
    }
    // kr rows scaled by value_inner_decay (compute staging)
    for (int i = 0; i < 32; ++i) {
      int ei = tid + 256 * i;             // 0..8191
      int cr = slab * 64 + (ei >> 7), col = ei & 127;
      unsigned short kk = 0;
      if (cr < 500) {
        float vid = expf(dh * (499.f - (float)cr)) / lrs;
        kk = f2bf(bf2f(krc[(size_t)cr * 128 + col]) * vid);
      }
      krsh[ei] = kk;
    }
    wait_async0();
    __syncthreads();
    #pragma unroll
    for (int ks = 0; ks < 2; ++ks) {
      bf16x16 a = loadA_strided(krsh, m0 + ll, 32 * ks, half, 128);
      #pragma unroll
      for (int nt = 0; nt < 8; ++nt) {
        bf16x16 b = loadB_strided(vsh, nt * 16 + ll, 32 * ks, half, 128);
        acc[nt] = wmma_bf16(a, b, acc[nt]);
      }
    }
  }
  float* out = kvb + (size_t)blk * 16384;
  #pragma unroll
  for (int nt = 0; nt < 8; ++nt)
    #pragma unroll
    for (int r = 0; r < 8; ++r)
      out[(m0 + r + 8 * half) * 128 + nt * 16 + ll] = acc[nt][r];
}

// ---------------- K5: 8-chunk scan of the KV state ----------------
__global__ __launch_bounds__(256) void k_scan(const float* kvb, unsigned short* sbf,
                                              float* csc) {
  int bh = blockIdx.x, h = bh & 7, tid = threadIdx.x;
  float dh = decay_log(h);
  float cd = expf(dh * 500.f);
  float st[64];
  #pragma unroll
  for (int i = 0; i < 64; ++i) st[i] = 0.f;
  __shared__ float colsum[128];
  __shared__ float csSh;
  float cs = 1.f;
  for (int n = 0; n < 8; ++n) {
    size_t base = ((size_t)bh * 8 + n) * 16384;
    #pragma unroll
    for (int i = 0; i < 64; ++i) {       // write state entering chunk n
      int e = tid + 256 * i;
      sbf[base + e] = f2bf(st[i]);
    }
    if (tid == 0) csc[bh * 8 + n] = cs;
    if (tid < 128) colsum[tid] = 0.f;
    __syncthreads();
    #pragma unroll
    for (int i = 0; i < 64; ++i) {
      int e = tid + 256 * i;
      st[i] = st[i] * cd + kvb[base + e];
      atomicAdd(&colsum[e & 127], fabsf(st[i]));
    }
    __syncthreads();
    if (tid == 0) {
      float mx = 0.f;
      for (int j = 0; j < 128; ++j) mx = fmaxf(mx, colsum[j]);
      csSh = fmaxf(mx, 1.f);
    }
    __syncthreads();
    cs = csSh;
  }
}

// ---------------- K6: retention main (qk + inner + cross, all WMMA) ----------------
__global__ __launch_bounds__(256) void k_ret(const unsigned short* qrb,
                                             const unsigned short* krb,
                                             const unsigned short* vbb,
                                             const unsigned short* sbf,
                                             const float* csc, float* ob) {
  int blk = blockIdx.x;                 // bh*8 + n
  int n = blk & 7, bh = blk >> 3, h = bh & 7, b = bh >> 3;
  int tid = threadIdx.x, wave = tid >> 5, lane = tid & 31;
  int half = lane >> 4, ll = lane & 15;
  float dh = decay_log(h);
  float em1 = 1.f - expf(dh);
  float lrs = (1.f - expf(dh * 500.f)) / em1;
  const unsigned short* qrc = qrb + ((size_t)bh * 4000 + n * 500) * 128;
  const unsigned short* krc = krb + ((size_t)bh * 4000 + n * 500) * 128;
  const unsigned short* vc  = vbb + ((size_t)bh * 4000 + n * 500) * 128;
  const unsigned short* Sn  = sbf + (size_t)blk * 16384;
  float crossS = csc[blk];
  __shared__ unsigned short vsh[128 * 128];       // 32KB v slab (async staged)
  __shared__ unsigned short tile[8][16 * 32];     // per-wave qk tile (bf16)
  __shared__ float iscale[8][16];                 // per-wave row |qk| sums

  for (int iter = 0; iter < 4; ++iter) {
    int c0 = (wave + 8 * iter) * 16;
    int mrow = c0 + ll;
    bf16x16 qa[4];
    #pragma unroll
    for (int ks = 0; ks < 4; ++ks)
      qa[ks] = (mrow < 500) ? loadA_row(qrc + (size_t)mrow * 128, 32 * ks, half)
                            : zfrag();
    // cross phase: acc = qr @ S_raw, then row-scale by qid
    f32x8 acc[8];
    #pragma unroll
    for (int nt = 0; nt < 8; ++nt) {
      f32x8 a = fzero();
      #pragma unroll
      for (int ks = 0; ks < 4; ++ks) {
        bf16x16 bb = loadB_strided(Sn, nt * 16 + ll, 32 * ks, half, 128);
        a = wmma_bf16(qa[ks], bb, a);
      }
      acc[nt] = a;
    }
    float qidv[8];
    #pragma unroll
    for (int r = 0; r < 8; ++r) {
      int c = c0 + r + 8 * half;
      float er = expf(dh * ((float)c + 1.f));
      float msum = (1.f - er) / em1;             // sum_{e<=c} r^(c-e)
      qidv[r] = er * lrs * rsqrtf(fmaxf(msum, 1e-30f));
    }
    #pragma unroll
    for (int nt = 0; nt < 8; ++nt) {
      #pragma unroll
      for (int r = 0; r < 8; ++r) acc[nt][r] *= qidv[r];
    }
    if (lane < 16) iscale[wave][lane] = 0.f;

    // inner phase over 4 v slabs of 128 chunk-rows
    for (int slab = 0; slab < 4; ++slab) {
      __syncthreads();
      // async DMA v slab to LDS (2048 x 16B chunks over 256 threads)
      #pragma unroll
      for (int i = 0; i < 8; ++i) {
        int c16 = tid + 256 * i;              // 0..2047
        int lr = c16 >> 4, cc = (c16 & 15) * 8;
        int cr = slab * 128 + lr;
        if (cr < 500)
          async_g2l_b128(&vsh[lr * 128 + cc], vc + (size_t)cr * 128 + cc);
        else
          zero16_lds(&vsh[lr * 128 + cc]);
      }
      wait_async0();
      __syncthreads();
      for (int sub = 0; sub < 4; ++sub) {
        int ek = slab * 4 + sub;               // 32-wide e block
        #pragma unroll
        for (int ht = 0; ht < 2; ++ht) {
          int ebase = ek * 32 + ht * 16;
          f32x8 qacc = fzero();
          int erow = ebase + ll;
          #pragma unroll
          for (int ks = 0; ks < 4; ++ks) {
            bf16x16 bb = (erow < 500)
                             ? loadB_row(krc + (size_t)erow * 128, 32 * ks, half)
                             : zfrag();
            qacc = wmma_bf16(qa[ks], bb, qacc);
          }
          #pragma unroll
          for (int r = 0; r < 8; ++r) {
            int c = c0 + r + 8 * half;
            int e = ebase + ll;
            float mval = 0.f;
            if (c < 500 && e <= c) {
              float er2 = expf(dh * ((float)c + 1.f));
              float msum = (1.f - er2) / em1;
              mval = expf(dh * (float)(c - e)) * rsqrtf(fmaxf(msum, 1e-30f));
            }
            float val = qacc[r] * mval;
            tile[wave][(r + 8 * half) * 32 + ht * 16 + ll] = f2bf(val);
            atomicAdd(&iscale[wave][r + 8 * half], fabsf(val));
          }
        }
        // transpose C-layout tile -> A fragment via LDS, accumulate qk @ v
        bf16x16 aq = loadA_row(&tile[wave][ll * 32], 0, half);
        #pragma unroll
        for (int nt = 0; nt < 8; ++nt) {
          bf16x16 bb = loadB_strided(vsh, nt * 16 + ll, 32 * sub, half, 128);
          acc[nt] = wmma_bf16(aq, bb, acc[nt]);
        }
      }
    }
    // finalize: divide by all_scale = max(inner_scale[c], cross_scale)
    #pragma unroll
    for (int nt = 0; nt < 8; ++nt) {
      #pragma unroll
      for (int r = 0; r < 8; ++r) {
        int c = c0 + r + 8 * half;
        if (c < 500) {
          float isc = fmaxf(iscale[wave][r + 8 * half], 1.f);
          float als = fmaxf(isc, crossS);
          ob[((size_t)b * 4000 + n * 500 + c) * 1024 + h * 128 + nt * 16 + ll] =
              acc[nt][r] / als;
        }
      }
    }
  }
}

// ---------------- K7: per-head RMS norm + silu(g) gate -> z bf16 ----------------
__global__ __launch_bounds__(256) void k_rmssilu(const float* ob, const float* gf,
                                                 unsigned short* zb) {
  int row = blockIdx.x, tid = threadIdx.x;
  const float* orow = ob + (size_t)row * 1024;
  const float* grow = gf + (size_t)row * 1024;
  float v[4];
  #pragma unroll
  for (int i = 0; i < 4; ++i) v[i] = orow[tid * 4 + i];
  float ss = v[0] * v[0] + v[1] * v[1] + v[2] * v[2] + v[3] * v[3];
  __shared__ float red[256];
  red[tid] = ss;
  __syncthreads();
  for (int off = 16; off; off >>= 1) {        // reduce within each head (32 thr)
    if ((tid & 31) < off) red[tid] += red[tid + off];
    __syncthreads();
  }
  float rn = rsqrtf(red[tid & ~31] * (1.f / 128.f) + 1e-6f);
  #pragma unroll
  for (int i = 0; i < 4; ++i) {
    int d = tid * 4 + i;
    float g = grow[d];
    float sig = 1.f / (1.f + expf(-g));
    zb[(size_t)row * 1024 + d] = f2bf(v[i] * rn * g * sig);
  }
}

// ---------------- K8: output GEMM z @ Wo (WMMA) ----------------
__global__ __launch_bounds__(256) void k_gemm_out(const unsigned short* zb,
                                                  const unsigned short* wto,
                                                  float* out) {
  int wave = threadIdx.x >> 5, lane = threadIdx.x & 31;
  int half = lane >> 4, ll = lane & 15;
  int row0 = blockIdx.x * 128 + wave * 16;
  const unsigned short* arow = zb + (size_t)(row0 + ll) * 1024;
  const unsigned short* wbase = wto + ((size_t)blockIdx.y * 128) * 1024;
  f32x8 acc[8];
  #pragma unroll
  for (int i = 0; i < 8; ++i) acc[i] = fzero();
  for (int k0 = 0; k0 < 1024; k0 += 32) {
    if (k0 + 64 < 1024) {
      __builtin_prefetch(arow + k0 + 64, 0, 3);
      __builtin_prefetch(wbase + (size_t)ll * 1024 + k0 + 64, 0, 3);
    }
    bf16x16 a = loadA_row(arow, k0, half);
    #pragma unroll
    for (int nt = 0; nt < 8; ++nt) {
      bf16x16 b = loadB_row(wbase + (size_t)(nt * 16 + ll) * 1024, k0, half);
      acc[nt] = wmma_bf16(a, b, acc[nt]);
    }
  }
  #pragma unroll
  for (int nt = 0; nt < 8; ++nt) {
    #pragma unroll
    for (int r = 0; r < 8; ++r) {
      int row = row0 + r + 8 * half;
      out[(size_t)row * 1024 + blockIdx.y * 128 + nt * 16 + ll] = acc[nt][r];
    }
  }
}

extern "C" void kernel_launch(void* const* d_in, const int* in_sizes, int n_in,
                              void* d_out, int out_size, void* d_ws, size_t ws_size,
                              hipStream_t stream) {
  (void)in_sizes; (void)n_in; (void)out_size; (void)ws_size;
  const float* x_in = (const float*)d_in[0];
  const float* gma  = (const float*)d_in[1];
  const float* bta  = (const float*)d_in[2];
  const float* Wq   = (const float*)d_in[3];
  const float* Wk   = (const float*)d_in[4];
  const float* Wv   = (const float*)d_in[5];
  const float* Wg   = (const float*)d_in[6];
  const float* Wo   = (const float*)d_in[7];
  float* out = (float*)d_out;

  char* ws = (char*)d_ws;
  size_t off = 0;
  auto alloc = [&](size_t bytes) -> void* {
    off = (off + 255) & ~(size_t)255;
    void* p = ws + off;
    off += bytes;
    return p;
  };
  const size_t BT = 4 * 4000;            // 16000
  unsigned short* xb  = (unsigned short*)alloc(BT * 1024 * 2);
  unsigned short* wt  = (unsigned short*)alloc((size_t)5 * 1024 * 1024 * 2);
  float*          qf  = (float*)alloc(BT * 1024 * 4);
  float*          kf  = (float*)alloc(BT * 1024 * 4);
  unsigned short* vbb = (unsigned short*)alloc(BT * 1024 * 2);
  float*          gf  = (float*)alloc(BT * 1024 * 4);
  unsigned short* qrb = (unsigned short*)alloc(BT * 1024 * 2);
  unsigned short* krb = (unsigned short*)alloc(BT * 1024 * 2);
  float*          kvb = (float*)alloc((size_t)4 * 8 * 8 * 16384 * 4);
  unsigned short* sbf = (unsigned short*)alloc((size_t)4 * 8 * 8 * 16384 * 2);
  float*          csc = (float*)alloc((size_t)4 * 8 * 8 * 4);
  float*          ob  = (float*)alloc(BT * 1024 * 4);
  unsigned short* zb  = (unsigned short*)alloc(BT * 1024 * 2);

  k_prep_w<<<20480, 256, 0, stream>>>(Wq, Wk, Wv, Wg, Wo, wt);
  k_ln<<<16000, 256, 0, stream>>>(x_in, gma, bta, xb);
  k_gemm_qkvg<<<dim3(125, 32), 256, 0, stream>>>(xb, wt, qf, kf, vbb, gf);
  k_theta<<<32000, 256, 0, stream>>>(qf, kf, qrb, krb);
  k_kv<<<256, 256, 0, stream>>>(krb, vbb, kvb);
  k_scan<<<32, 256, 0, stream>>>(kvb, sbf, csc);
  k_ret<<<256, 256, 0, stream>>>(qrb, krb, vbb, sbf, csc, ob);
  k_rmssilu<<<16000, 256, 0, stream>>>(ob, gf, zb);
  k_gemm_out<<<dim3(125, 8), 256, 0, stream>>>(zb, wt + (size_t)4 * 1024 * 1024, out);
}
